// GCN_8143257993640
// MI455X (gfx1250) — matrix operand order repaired and checked by reading, not compiled, not run on hardware
//
#include <hip/hip_runtime.h>
#include <hip/hip_bf16.h>

#define NN 16384
#define DIM 64
#define BN_EPS 1e-5f
#define SLOPE 0.01f

typedef __attribute__((ext_vector_type(16))) __bf16 v16bf;
typedef __attribute__((ext_vector_type(8)))  float  v8f;
typedef __attribute__((ext_vector_type(4)))  float  v4f;

// ---------------- BN statistics: per-column sum / sum-of-squares ----------------
__global__ void k_bn_stats(const float* __restrict__ h,
                           float* __restrict__ colsum, float* __restrict__ colsq) {
    __shared__ float ss[256], sq[256];
    int t = threadIdx.x;
    int c = t & 63;              // column
    int rg = t >> 6;             // 0..3 row-group
    int rbase = blockIdx.x * 256;
    float s = 0.f, q = 0.f;
    for (int r = rbase + rg; r < rbase + 256; r += 4) {
        float v = h[(size_t)r * DIM + c];
        s += v; q += v * v;
    }
    ss[t] = s; sq[t] = q;
    __syncthreads();
    if (t < 64) {
        s = ss[t] + ss[t + 64] + ss[t + 128] + ss[t + 192];
        q = sq[t] + sq[t + 64] + sq[t + 128] + sq[t + 192];
        atomicAdd(&colsum[t], s);
        atomicAdd(&colsq[t], q);
    }
}

__global__ void k_bn_finalize(const float* __restrict__ colsum, const float* __restrict__ colsq,
                              const float* __restrict__ gamma, const float* __restrict__ beta,
                              float* __restrict__ scale, float* __restrict__ shift) {
    int c = threadIdx.x;
    float mean = colsum[c] * (1.0f / NN);
    float var  = colsq[c] * (1.0f / NN) - mean * mean;   // biased variance
    float sc   = gamma[c] * rsqrtf(var + BN_EPS);
    scale[c] = sc;
    shift[c] = beta[c] - mean * sc;
}

// ---------------- row sums of (g + I) -> dinv = rsqrt(deg) ----------------
__global__ void k_rowsum(const float* __restrict__ g, float* __restrict__ dinv) {
    __shared__ float red[256];
    int row = blockIdx.x;
    const v4f* gp = (const v4f*)(g + (size_t)row * NN);
    float s = 0.f;
    for (int i = threadIdx.x; i < NN / 4; i += 256) {
        v4f v = __builtin_nontemporal_load(gp + i);   // stream-once: NT hint
        s += (v[0] + v[1]) + (v[2] + v[3]);
    }
    red[threadIdx.x] = s;
    __syncthreads();
    for (int off = 128; off > 0; off >>= 1) {
        if (threadIdx.x < off) red[threadIdx.x] += red[threadIdx.x + off];
        __syncthreads();
    }
    if (threadIdx.x == 0) dinv[row] = rsqrtf(red[0] + 1.0f);  // +1 self loop
}

// ---------------- hp = (h*scale+shift) @ W^T + b ----------------
__global__ void k_hp(const float* __restrict__ h, const float* __restrict__ W,
                     const float* __restrict__ b, const float* __restrict__ scale,
                     const float* __restrict__ shift, float* __restrict__ hp) {
    __shared__ float hn[64];
    int r = blockIdx.x, t = threadIdx.x;
    hn[t] = h[(size_t)r * DIM + t] * scale[t] + shift[t];
    __syncthreads();
    float acc = b[t];
#pragma unroll
    for (int k = 0; k < 64; ++k) acc += hn[k] * W[t * 64 + k];
    hp[(size_t)r * DIM + t] = acc;
}

// ---------------- hq = dinv .* hp, stored TRANSPOSED as bf16 hi+lo ----------------
__global__ void k_hq(const float* __restrict__ hp, const float* __restrict__ dinv,
                     __bf16* __restrict__ qhi, __bf16* __restrict__ qlo) {
    int j = blockIdx.x * 256 + threadIdx.x;   // node
    int c = blockIdx.y;                       // feature
    float v = dinv[j] * hp[(size_t)j * DIM + c];
    __bf16 hi = (__bf16)v;
    qhi[(size_t)c * NN + j] = hi;
    qlo[(size_t)c * NN + j] = (__bf16)(v - (float)hi);
}

// ---------------- main propagation: out = leaky( dinv .* ((g+I) @ hq) ) ----------------
// One wave owns a 16-row stripe and all 64 output columns (4 WMMA N-tiles).
// Split-precision bf16 (hi+lo, 3 WMMA terms) ~= f32 accuracy while staying
// memory-bound on the 1.07 GB g stream. WMMAs are issued term-major so that
// dependent WMMAs on the same accumulator are separated by 3 independent ones
// (covers the WMMA->WMMA RAW hazard without v_nop padding).
__global__ __launch_bounds__(128) void k_prop(const float* __restrict__ g,
                                              const __bf16* __restrict__ qhi,
                                              const __bf16* __restrict__ qlo,
                                              const float* __restrict__ hp,
                                              const float* __restrict__ dinv,
                                              float* __restrict__ out) {
    int wave = blockIdx.x * (blockDim.x >> 5) + (threadIdx.x >> 5);
    int lane = threadIdx.x & 31;
    int lrow = lane & 15;     // M within tile (A rows), N within tile (B/C cols)
    int half = lane >> 4;     // lane half selects K / M sub-range per ISA layout
    int r0 = wave * 16;

    const float* gp = g + (size_t)(r0 + lrow) * NN;

    v8f acc0 = {}, acc1 = {}, acc2 = {}, acc3 = {};

    union BU { uint4 q[2]; v16bf v; };

    for (int k0 = 0; k0 < NN; k0 += 32) {
        // ---- A tile: 16x32 f32 of g -> bf16 hi/lo, contiguous-K A layout (same K order as B) ----
        // lane half 0: K = k0+[0..15]; half 1: K = k0+[16..31].
        int base = k0 + half * 16;
        v4f a0 = __builtin_nontemporal_load((const v4f*)(gp + base));
        v4f a1 = __builtin_nontemporal_load((const v4f*)(gp + base + 4));
        v4f a2 = __builtin_nontemporal_load((const v4f*)(gp + base + 8));
        v4f a3 = __builtin_nontemporal_load((const v4f*)(gp + base + 12));
        __builtin_prefetch(gp + k0 + 64, 0, 0);   // stream g ahead

        float af[16] = {a0[0], a0[1], a0[2], a0[3], a1[0], a1[1], a1[2], a1[3],
                        a2[0], a2[1], a2[2], a2[3], a3[0], a3[1], a3[2], a3[3]};
        v16bf ahi, alo;
#pragma unroll
        for (int e = 0; e < 16; ++e) {
            float f = af[e];
            __bf16 hi = (__bf16)f;            // v_cvt_pk_bf16_f32 (RNE)
            ahi[e] = hi;
            alo[e] = (__bf16)(f - (float)hi); // residual via v_fma_mix_f32_bf16
        }

        // ---- B tiles (all 4, hi+lo) from transposed hq: 16 consecutive K per lane ----
        size_t boff = (size_t)lrow * NN + (size_t)k0 + (size_t)half * 16;
        BU bh[4], bl[4];
#pragma unroll
        for (int n = 0; n < 4; ++n) {
            const __bf16* ph = qhi + (size_t)(n * 16) * NN + boff;
            const __bf16* pl = qlo + (size_t)(n * 16) * NN + boff;
            bh[n].q[0] = *(const uint4*)(ph);
            bh[n].q[1] = *(const uint4*)(ph + 8);
            bl[n].q[0] = *(const uint4*)(pl);
            bl[n].q[1] = *(const uint4*)(pl + 8);
        }

        // ---- term-major WMMA issue: round-robin over accumulators ----
        acc0 = __builtin_amdgcn_wmma_f32_16x16x32_bf16(false, ahi, false, bh[0].v, (short)0, acc0, false, false);
        acc1 = __builtin_amdgcn_wmma_f32_16x16x32_bf16(false, ahi, false, bh[1].v, (short)0, acc1, false, false);
        acc2 = __builtin_amdgcn_wmma_f32_16x16x32_bf16(false, ahi, false, bh[2].v, (short)0, acc2, false, false);
        acc3 = __builtin_amdgcn_wmma_f32_16x16x32_bf16(false, ahi, false, bh[3].v, (short)0, acc3, false, false);

        acc0 = __builtin_amdgcn_wmma_f32_16x16x32_bf16(false, ahi, false, bl[0].v, (short)0, acc0, false, false);
        acc1 = __builtin_amdgcn_wmma_f32_16x16x32_bf16(false, ahi, false, bl[1].v, (short)0, acc1, false, false);
        acc2 = __builtin_amdgcn_wmma_f32_16x16x32_bf16(false, ahi, false, bl[2].v, (short)0, acc2, false, false);
        acc3 = __builtin_amdgcn_wmma_f32_16x16x32_bf16(false, ahi, false, bl[3].v, (short)0, acc3, false, false);

        acc0 = __builtin_amdgcn_wmma_f32_16x16x32_bf16(false, alo, false, bh[0].v, (short)0, acc0, false, false);
        acc1 = __builtin_amdgcn_wmma_f32_16x16x32_bf16(false, alo, false, bh[1].v, (short)0, acc1, false, false);
        acc2 = __builtin_amdgcn_wmma_f32_16x16x32_bf16(false, alo, false, bh[2].v, (short)0, acc2, false, false);
        acc3 = __builtin_amdgcn_wmma_f32_16x16x32_bf16(false, alo, false, bh[3].v, (short)0, acc3, false, false);
    }

    // ---- epilogue: add self-loop term, scale by dinv[i], LeakyReLU ----
    // C layout: VGPR r -> row r0 + r + half*8; lane%16 -> column within N-tile.
#pragma unroll
    for (int n = 0; n < 4; ++n) {
        v8f a = (n == 0) ? acc0 : (n == 1) ? acc1 : (n == 2) ? acc2 : acc3;
        int col = n * 16 + lrow;
#pragma unroll
        for (int r = 0; r < 8; ++r) {
            int row = r0 + r + half * 8;
            float di = dinv[row];
            float diag = di * hp[(size_t)row * DIM + col];  // I @ hq term, exact f32
            float v = di * (a[r] + diag);
            out[(size_t)row * DIM + col] = (v >= 0.f) ? v : SLOPE * v;
        }
    }
}

extern "C" void kernel_launch(void* const* d_in, const int* in_sizes, int n_in,
                              void* d_out, int out_size, void* d_ws, size_t ws_size,
                              hipStream_t stream) {
    const float* g     = (const float*)d_in[0];
    const float* h     = (const float*)d_in[1];
    const float* W     = (const float*)d_in[2];
    const float* b     = (const float*)d_in[3];
    const float* gamma = (const float*)d_in[4];
    const float* beta  = (const float*)d_in[5];
    float* out = (float*)d_out;

    // workspace carve-up (~8.5 MB)
    float* colsum = (float*)d_ws;                 // 64
    float* colsq  = colsum + 64;                  // 64
    float* scale  = colsum + 128;                 // 64
    float* shift  = colsum + 192;                 // 64
    float* dinv   = colsum + 256;                 // NN
    float* hp     = dinv + NN;                    // NN*64 f32
    __bf16* qhi = (__bf16*)(hp + (size_t)NN * DIM);   // 64 x NN bf16 (transposed)
    __bf16* qlo = qhi + (size_t)NN * DIM;             // 64 x NN bf16 (transposed)

    hipMemsetAsync(colsum, 0, 2 * 64 * sizeof(float), stream);

    k_bn_stats   <<<NN / 256, 256, 0, stream>>>(h, colsum, colsq);
    k_bn_finalize<<<1, 64, 0, stream>>>(colsum, colsq, gamma, beta, scale, shift);
    k_rowsum     <<<NN, 256, 0, stream>>>(g, dinv);
    k_hp         <<<NN, 64, 0, stream>>>(h, W, b, scale, shift, hp);
    dim3 gq(NN / 256, DIM);
    k_hq         <<<gq, 256, 0, stream>>>(hp, dinv, qhi, qlo);
    // 1024 row-tiles, 4 waves (128 threads) per block
    k_prop       <<<NN / 16 / 4, 128, 0, stream>>>(g, qhi, qlo, hp, dinv, out);
}
